// TransformerBlock_1932735283786
// MI455X (gfx1250) — matrix operand order repaired and checked by reading, not compiled, not run on hardware
//
#include <hip/hip_runtime.h>
#include <hip/hip_bf16.h>

typedef unsigned short u16;
typedef __attribute__((ext_vector_type(16))) __bf16 v16bf;
typedef __attribute__((ext_vector_type(8)))  float  v8f;

union Frag {
    v16bf v;
    uint4 q[2];
};

__device__ __forceinline__ u16 f2bf(float f) {
    union { float f; unsigned u; } x;
    x.f = f;
    unsigned r = x.u + 0x7FFFu + ((x.u >> 16) & 1u);
    return (u16)(r >> 16);
}

// A-operand fragment (16x32 bf16, MxK): lane L: row = L&15,
// K = {8*hi .. 8*hi+7} U {8*hi+16 .. 8*hi+23}   (hi = L>>4)
__device__ __forceinline__ Frag load_a_frag(const u16* base, int stride, int r, int hi) {
    Frag f;
    const u16* p = base + (size_t)r * stride + hi * 8;
    f.q[0] = *(const uint4*)p;
    f.q[1] = *(const uint4*)(p + 16);
    return f;
}

// B-operand fragment (32x16 bf16, KxN) from row-major [N,K] source:
// lane L: col n = L&15, K = {8*hi .. 8*hi+7} U {8*hi+16 .. 8*hi+23}.
__device__ __forceinline__ Frag load_b_frag(const u16* base, int stride, int r, int hi) {
    Frag f;
    const u16* p = base + (size_t)r * stride + hi * 8;
    f.q[0] = *(const uint4*)p;
    f.q[1] = *(const uint4*)(p + 16);
    return f;
}

#define WMMA_BF16(a, b, c) \
    __builtin_amdgcn_wmma_f32_16x16x32_bf16(false, (a), false, (b), (short)0, (c), false, false)

// CDNA5 async global->LDS copy (ASYNCcnt path). LDS address = low 32 bits of
// the flat shared pointer.
__device__ __forceinline__ void async_load_b128(void* lds_ptr, const void* gptr) {
    unsigned lds_off = (unsigned)(unsigned long long)lds_ptr;
    unsigned long long ga = (unsigned long long)gptr;
    asm volatile("global_load_async_to_lds_b128 %0, %1, off"
                 :: "v"(lds_off), "v"(ga)
                 : "memory");
}
#define ASYNC_WAIT_8() asm volatile("s_wait_asynccnt 8" ::: "memory")
#define ASYNC_WAIT_0() asm volatile("s_wait_asynccnt 0" ::: "memory")

__device__ __forceinline__ float red_max16(float x) {
#pragma unroll
    for (int m = 1; m <= 8; m <<= 1) x = fmaxf(x, __shfl_xor(x, m, 32));
    return x;
}
__device__ __forceinline__ float red_sum16(float x) {
#pragma unroll
    for (int m = 1; m <= 8; m <<= 1) x += __shfl_xor(x, m, 32);
    return x;
}

// ---------------------------------------------------------------- f32 -> bf16
__global__ __launch_bounds__(256) void cvt_bf16_kernel(const float* __restrict__ in,
                                                       u16* __restrict__ out, int n) {
    int i = blockIdx.x * 256 + threadIdx.x;
    int stride = gridDim.x * 256;
    for (; i < n; i += stride) out[i] = f2bf(in[i]);
}

// ---------------------------------------------------------------- LayerNorm (C=1024) -> bf16
__global__ __launch_bounds__(256) void ln_bf16_kernel(const float* __restrict__ x,
                                                      const float* __restrict__ g,
                                                      const float* __restrict__ bta,
                                                      u16* __restrict__ out) {
    const int row = blockIdx.x, tid = threadIdx.x;
    const float* xr = x + (size_t)row * 1024;
    float v[4];
    float s = 0.f, s2 = 0.f;
#pragma unroll
    for (int i = 0; i < 4; ++i) {
        float t = xr[tid + 256 * i];
        v[i] = t; s += t; s2 += t * t;
    }
#pragma unroll
    for (int m = 1; m < 32; m <<= 1) {
        s  += __shfl_xor(s,  m, 32);
        s2 += __shfl_xor(s2, m, 32);
    }
    __shared__ float ps[8], ps2[8];
    if ((tid & 31) == 0) { ps[tid >> 5] = s; ps2[tid >> 5] = s2; }
    __syncthreads();
    float ts = 0.f, ts2 = 0.f;
#pragma unroll
    for (int w = 0; w < 8; ++w) { ts += ps[w]; ts2 += ps2[w]; }
    const float mu  = ts * (1.0f / 1024.0f);
    const float var = ts2 * (1.0f / 1024.0f) - mu * mu;
    const float rs  = rsqrtf(var + 1e-5f);
    u16* orow = out + (size_t)row * 1024;
#pragma unroll
    for (int i = 0; i < 4; ++i) {
        int c = tid + 256 * i;
        orow[c] = f2bf((v[i] - mu) * rs * g[c] + bta[c]);
    }
}

// ---------------------------------------------------------------- generic WMMA GEMM
// out[m,n] = epilogue( sum_k A[m,k]*W[n,k] + bias[n] (+ resid[m,n]) )
// BM=128, BN=128, BK=64; 256 threads = 8 waves; wave computes 64x32.
// Double-buffered LDS fed by async global->LDS DMA (ASYNCcnt).
template <bool GELU, bool RESID, bool OUTBF>
__global__ __launch_bounds__(256) void gemm_bf16_kernel(const u16* __restrict__ A,
                                                        const u16* __restrict__ W,
                                                        const float* __restrict__ bias,
                                                        const float* __restrict__ resid,
                                                        float* __restrict__ outf,
                                                        u16* __restrict__ outb,
                                                        int M, int N, int K) {
    __shared__ u16 As[2][128 * 64];
    __shared__ u16 Bs[2][128 * 64];
    const int tid  = threadIdx.x;
    const int lane = tid & 31, wv = tid >> 5;
    const int r = lane & 15, hi = lane >> 4;
    const int wm = wv & 1;            // 2 waves over M (64 rows each)
    const int wn = wv >> 1;           // 4 waves over N (32 cols each)
    const int bm = blockIdx.y * 128, bn = blockIdx.x * 128;

    // this thread's 4 tile chunks (8 bf16 each) per operand
    const int lrow = tid >> 3;               // chunk row for i=0 (rows advance by 32 per i)
    const int lc8  = (tid & 7) << 3;         // chunk col

    v8f acc[4][2];
    v8f vz = {};
#pragma unroll
    for (int i = 0; i < 4; ++i)
#pragma unroll
        for (int j = 0; j < 2; ++j) acc[i][j] = vz;

    const int nkt = K >> 6;

    // prologue: tile 0 -> buffer 0 (8 async wave-instructions)
#pragma unroll
    for (int i = 0; i < 4; ++i) {
        int row = lrow + 32 * i;
        async_load_b128(&As[0][row * 64 + lc8], &A[(size_t)(bm + row) * K + lc8]);
        async_load_b128(&Bs[0][row * 64 + lc8], &W[(size_t)(bn + row) * K + lc8]);
    }

    for (int kt = 0; kt < nkt; ++kt) {
        const int cur = kt & 1;
        if (kt + 1 < nkt) {
            // stream next tile while current computes (WAR safe: end-of-iter
            // barrier of kt-1 guarantees everyone is done reading this buffer)
#pragma unroll
            for (int i = 0; i < 4; ++i) {
                int row = lrow + 32 * i;
                async_load_b128(&As[cur ^ 1][row * 64 + lc8],
                                &A[(size_t)(bm + row) * K + (kt + 1) * 64 + lc8]);
                async_load_b128(&Bs[cur ^ 1][row * 64 + lc8],
                                &W[(size_t)(bn + row) * K + (kt + 1) * 64 + lc8]);
            }
            ASYNC_WAIT_8();     // async ops retire in order: tile kt's 8 are done
        } else {
            ASYNC_WAIT_0();
        }
        __syncthreads();        // all waves' tile-kt DMA visible in LDS

#pragma unroll
        for (int ks = 0; ks < 2; ++ks) {
            Frag af[4], bf[2];
#pragma unroll
            for (int mt = 0; mt < 4; ++mt)
                af[mt] = load_a_frag(&As[cur][(wm * 64 + mt * 16) * 64 + ks * 32], 64, r, hi);
#pragma unroll
            for (int nt = 0; nt < 2; ++nt)
                bf[nt] = load_b_frag(&Bs[cur][(wn * 32 + nt * 16) * 64 + ks * 32], 64, r, hi);
#pragma unroll
            for (int mt = 0; mt < 4; ++mt)
#pragma unroll
                for (int nt = 0; nt < 2; ++nt)
                    acc[mt][nt] = WMMA_BF16(af[mt].v, bf[nt].v, acc[mt][nt]);
        }
        __syncthreads();        // everyone done reading buf[cur] before it is re-filled
    }

#pragma unroll
    for (int mt = 0; mt < 4; ++mt)
#pragma unroll
        for (int nt = 0; nt < 2; ++nt)
#pragma unroll
            for (int v = 0; v < 8; ++v) {
                int row = bm + wm * 64 + mt * 16 + v + hi * 8;
                int col = bn + wn * 32 + nt * 16 + r;
                float val = acc[mt][nt][v] + bias[col];
                if constexpr (RESID) val += resid[(size_t)row * N + col];
                if constexpr (GELU)  val = 0.5f * val * (1.0f + erff(val * 0.70710678118654752f));
                if constexpr (OUTBF) outb[(size_t)row * N + col] = f2bf(val);
                else                 outf[(size_t)row * N + col] = val;
            }
}

// ---------------------------------------------------------------- v-transpose: qkv -> vT[b,h,d,n]
__global__ __launch_bounds__(256) void vtrans_kernel(const u16* __restrict__ qkv,
                                                     u16* __restrict__ vT) {
    int idx = blockIdx.x * 256 + threadIdx.x;      // 4194304 total
    int n = idx & 2047;
    int d = (idx >> 11) & 63;
    int bh = idx >> 17;
    int h = bh & 15, b = bh >> 4;
    vT[idx] = qkv[((size_t)(b * 2048 + n)) * 3072 + 2048 + h * 64 + d];
}

// ---------------------------------------------------------------- flash attention
// One wave per 16-row query tile; keys processed 32 at a time.
__global__ __launch_bounds__(256) void attn_kernel(const u16* __restrict__ qkv,
                                                   const u16* __restrict__ vT,
                                                   u16* __restrict__ o) {
    __shared__ u16 P[8 * 16 * 40];
    const int tid = threadIdx.x, wave = tid >> 5, lane = tid & 31;
    const int r = lane & 15, hi = lane >> 4;
    const int bh = blockIdx.y, b = bh >> 4, h = bh & 15;
    const int qt = blockIdx.x * 8 + wave;
    u16* Pw = &P[wave * 16 * 40];

    Frag aq[2];
#pragma unroll
    for (int ks = 0; ks < 2; ++ks)
        aq[ks] = load_a_frag(qkv + (size_t)(b * 2048 + qt * 16) * 3072 + h * 64 + ks * 32,
                             3072, r, hi);

    v8f vz = {};
    v8f oacc[4];
#pragma unroll
    for (int t = 0; t < 4; ++t) oacc[t] = vz;
    float mrun[8], lrun[8];
#pragma unroll
    for (int v = 0; v < 8; ++v) { mrun[v] = -1e30f; lrun[v] = 0.f; }

    const u16* kbase = qkv + (size_t)b * 2048 * 3072 + 1024 + h * 64;
    const u16* vbase = vT + (size_t)bh * 64 * 2048;

    for (int kb = 0; kb < 64; ++kb) {
        Frag bk[2][2];
#pragma unroll
        for (int c = 0; c < 2; ++c)
#pragma unroll
            for (int ks = 0; ks < 2; ++ks)
                bk[c][ks] = load_b_frag(kbase + (size_t)(kb * 32 + c * 16) * 3072 + ks * 32,
                                        3072, r, hi);
        v8f s0 = vz, s1 = vz;
        s0 = WMMA_BF16(aq[0].v, bk[0][0].v, s0);
        s0 = WMMA_BF16(aq[1].v, bk[0][1].v, s0);
        s1 = WMMA_BF16(aq[0].v, bk[1][0].v, s1);
        s1 = WMMA_BF16(aq[1].v, bk[1][1].v, s1);

        __asm volatile("" ::: "memory");
#pragma unroll
        for (int v = 0; v < 8; ++v) {
            float a0 = s0[v] * 0.125f, a1 = s1[v] * 0.125f;   // D^-0.5 = 1/8
            float mx = red_max16(fmaxf(a0, a1));
            float mn = fmaxf(mrun[v], mx);
            float e0 = __expf(a0 - mn), e1 = __expf(a1 - mn);
            float rs = red_sum16(e0 + e1);
            float al = __expf(mrun[v] - mn);
            lrun[v] = lrun[v] * al + rs;
            mrun[v] = mn;
#pragma unroll
            for (int t = 0; t < 4; ++t) oacc[t][v] *= al;
            // C-layout element (row = v + 8*hi, col = c*16 + r) -> LDS
            Pw[(v + hi * 8) * 40 + r]      = f2bf(e0);
            Pw[(v + hi * 8) * 40 + 16 + r] = f2bf(e1);
        }
        __asm volatile("" ::: "memory");

        Frag ap = load_a_frag(Pw, 40, r, hi);    // repacked P as A operand (16x32)
#pragma unroll
        for (int t = 0; t < 4; ++t) {
            Frag bv = load_b_frag(vbase + (size_t)(t * 16) * 2048 + kb * 32, 2048, r, hi);
            oacc[t] = WMMA_BF16(ap.v, bv.v, oacc[t]);
        }
        __asm volatile("" ::: "memory");
    }

#pragma unroll
    for (int v = 0; v < 8; ++v) {
        float inv = 1.0f / lrun[v];
#pragma unroll
        for (int t = 0; t < 4; ++t) {
            int row = b * 2048 + qt * 16 + v + hi * 8;
            int col = h * 64 + t * 16 + r;
            o[(size_t)row * 1024 + col] = f2bf(oacc[t][v] * inv);
        }
    }
}

// ================================================================ host launcher
extern "C" void kernel_launch(void* const* d_in, const int* in_sizes, int n_in,
                              void* d_out, int out_size, void* d_ws, size_t ws_size,
                              hipStream_t stream) {
    const float* x     = (const float*)d_in[0];
    const float* ln1g  = (const float*)d_in[1];
    const float* ln1b  = (const float*)d_in[2];
    const float* qkvw  = (const float*)d_in[3];
    const float* qkvbi = (const float*)d_in[4];
    const float* projw = (const float*)d_in[5];
    const float* projb = (const float*)d_in[6];
    const float* ln2g  = (const float*)d_in[7];
    const float* ln2b  = (const float*)d_in[8];
    const float* fc1w  = (const float*)d_in[9];
    const float* fc1b  = (const float*)d_in[10];
    const float* fc2w  = (const float*)d_in[11];
    const float* fc2b  = (const float*)d_in[12];
    float* out = (float*)d_out;

    const int M = 4096, C = 1024, Dff = 4096, NC3 = 3072;

    char* ws = (char*)d_ws;
    size_t off = 0;
    auto alloc = [&](size_t bytes) -> void* {
        void* p = ws + off;
        off += (bytes + 255) & ~(size_t)255;
        return p;
    };
    u16* wqkv  = (u16*)alloc((size_t)NC3 * C * 2);
    u16* wproj = (u16*)alloc((size_t)C * C * 2);
    u16* wfc1  = (u16*)alloc((size_t)Dff * C * 2);
    u16* wfc2  = (u16*)alloc((size_t)C * Dff * 2);
    u16* hb    = (u16*)alloc((size_t)M * C * 2);       // LN1 out; reused for LN2 out
    float* x2  = (float*)alloc((size_t)M * C * 4);     // residual after attention
    u16* ob    = (u16*)alloc((size_t)M * C * 2);       // attention output (bf16)
    u16* qkvb16 = (u16*)alloc((size_t)M * NC3 * 2);    // qkv (bf16)
    u16* vTb    = (u16*)alloc((size_t)M * C * 2);      // v transposed [b,h,d,n]
    u16* h2b = hb;                                     // LN2 out reuses hb
    u16* h3b = qkvb16;                                 // FC1/GELU out reuses qkv+vT slab (33.5MB)

    // 1) weights -> bf16
    cvt_bf16_kernel<<<2048, 256, 0, stream>>>(qkvw,  wqkv,  NC3 * C);
    cvt_bf16_kernel<<<2048, 256, 0, stream>>>(projw, wproj, C * C);
    cvt_bf16_kernel<<<2048, 256, 0, stream>>>(fc1w,  wfc1,  Dff * C);
    cvt_bf16_kernel<<<2048, 256, 0, stream>>>(fc2w,  wfc2,  C * Dff);

    // 2) LN1
    ln_bf16_kernel<<<M, 256, 0, stream>>>(x, ln1g, ln1b, hb);

    // 3) QKV GEMM: [4096,3072] = hb[4096,1024] @ qkv_w[3072,1024]^T + b
    gemm_bf16_kernel<false, false, true><<<dim3(NC3 / 128, M / 128), 256, 0, stream>>>(
        hb, wqkv, qkvbi, nullptr, nullptr, qkvb16, M, NC3, C);

    // 4) v transpose
    vtrans_kernel<<<(M * C) / 256, 256, 0, stream>>>(qkvb16, vTb);

    // 5) flash attention -> ob
    attn_kernel<<<dim3(16, 32), 256, 0, stream>>>(qkvb16, vTb, ob);

    // 6) proj GEMM + residual -> x2 (f32)
    gemm_bf16_kernel<false, true, false><<<dim3(C / 128, M / 128), 256, 0, stream>>>(
        ob, wproj, projb, x, x2, nullptr, M, C, C);

    // 7) LN2
    ln_bf16_kernel<<<M, 256, 0, stream>>>(x2, ln2g, ln2b, h2b);

    // 8) FC1 GEMM + GELU -> h3b (bf16)
    gemm_bf16_kernel<true, false, true><<<dim3(Dff / 128, M / 128), 256, 0, stream>>>(
        h2b, wfc1, fc1b, nullptr, nullptr, h3b, M, Dff, C);

    // 9) FC2 GEMM + residual -> out (f32)
    gemm_bf16_kernel<false, true, false><<<dim3(C / 128, M / 128), 256, 0, stream>>>(
        h3b, wfc2, fc2b, x2, out, nullptr, M, C, Dff);
}
